// EquivariantCNF_5841155523151
// MI455X (gfx1250) — compile-verified, matrix operand order checked by reading, not run on hardware
//
#include <hip/hip_runtime.h>
#include <hip/hip_bf16.h>
#include <math.h>

// ---------------------------------------------------------------------------
// EquivariantCNF forward, fused for MI455X (gfx1250, wave32, WMMA + TDM).
//
// All pairwise GEMMs are K=32 -> V_WMMA_F32_16X16X32_F16 (f16 in, f32 acc).
// Weights are pre-swizzled into the ISA B-operand lane layout; s / m_s are
// kept in the ISA A-operand lane layout (f16) so fragments load as single
// 32-byte vector loads. Edge/attention/coord per-pair work is fused with an
// online (flash) softmax so the (B,N,N,32) edge tensor never hits memory.
// hj tiles are streamed into LDS with the Tensor Data Mover (double
// buffered, TENSORcnt-tracked) so the copy overlaps the WMMA pipeline.
// ---------------------------------------------------------------------------

typedef __attribute__((ext_vector_type(16))) _Float16     v16h;
typedef __attribute__((ext_vector_type(8)))  float        v8f;
typedef __attribute__((ext_vector_type(4)))  unsigned int u32x4;
typedef __attribute__((ext_vector_type(8)))  unsigned int u32x8;

union V16H { v16h v; _Float16 e[16]; };
union V8F  { v8f  v; float    e[8];  };

#define BATCH 2
#define NNODE 512
#define TOTN  (BATCH * NNODE)   // 1024
#define SD    128
#define ED    32
#define NLAY  8
#define MAXD2 22500.0f          // 150^2

__device__ __forceinline__ float silu(float v) {
    return v / (1.0f + __expf(-v));
}

__device__ __forceinline__ v8f wmma16(v16h a, v16h b, v8f c) {
    // D = A(16x32 f16) * B(32x16 f16) + C(16x16 f32)
    return __builtin_amdgcn_wmma_f32_16x16x32_f16(
        /*neg_a=*/false, a, /*neg_b=*/false, b,
        /*c_mod=*/(short)0, c, /*reuse_a=*/false, /*reuse_b=*/false);
}

// K index held by (lane, half h) of a 16-bit A/B WMMA operand (ISA 7.12.2):
//   lane 0-15  -> K in {0..7, 16..23};  lane 16-31 -> K in {8..15, 24..31}
__device__ __forceinline__ int kmap(int h, int hih) {
    return ((h >> 3) << 4) | (hih << 3) | (h & 7);
}

// --------------------- Tensor Data Mover: 2KB row to LDS -------------------
// Builds a minimal user D# (1-row tile, 512 x 4-byte elements) and issues
// TENSOR_LOAD_TO_LDS (2-group form; VADDR2/VADDR3 = NULL -> <=2D tensor).
// Completion tracked with TENSORcnt (S_WAIT_TENSORCNT).
__device__ __forceinline__ void tdm_load_tile(unsigned lds_addr, const float* gptr) {
    unsigned long long ga = (unsigned long long)(uintptr_t)gptr;
    u32x4 g0;
    g0[0] = 1u;                                     // count=1, user descriptor
    g0[1] = lds_addr;                               // LDS byte address
    g0[2] = (unsigned)ga;                           // global_addr[31:0]
    g0[3] = (unsigned)((ga >> 32) & 0x01ffffffu)    // global_addr[56:32]
          | (2u << 30);                             // type = 2 ("image")
    const unsigned td0   = 1u << 20;                // tensor_dim0 (OOB bound)
    const unsigned tile0 = 512u;                    // tile_dim0: 512 elements
    u32x8 g1;
    g1[0] = 2u << 16;                               // data_size=4B, wg_mask=0
    g1[1] = (td0 & 0xffffu) << 16;                  // tensor_dim0 lo
    g1[2] = (td0 >> 16) | (1u << 16);               // tensor_dim0 hi | dim1 lo=1
    g1[3] = tile0 << 16;                            // dim1 hi=0 | tile_dim0
    g1[4] = 1u;                                     // tile_dim1=1, tile_dim2=0
    g1[5] = tile0;                                  // tensor_dim0_stride lo
    g1[6] = 0u;
    g1[7] = 0u;
    asm volatile("tensor_load_to_lds %0, %1" :: "s"(g0), "s"(g1) : "memory");
}

// --------------------------- weight pre-swizzle ----------------------------
// One wave per 32x16 B tile. Per-layer tile table (tile = 512 f16 = 1KB):
//   [ 0, 8)  edgeW1 rows   0..127 (hi):  c*2+t   (c = K/32 chunk, t = N/16)
//   [ 8,16)  edgeW1 rows 128..255 (hj)
//   [16,18)  edgeW2     [18,20) attW1    [20,22) coordW1
//   [22,62)  scalarW1 (160x128): 22 + c*8 + t
//   [62,94)  scalarW2 (128x128): 62 + c*8 + t
#define TILES_PER_LAYER 94

__global__ void k_prep_weights(const float* eW1, const float* eW2,
                               const float* aW1, const float* cW1,
                               const float* sW1, const float* sW2,
                               _Float16* wt) {
    int gid = blockIdx.x;
    int l = gid / TILES_PER_LAYER, tile = gid % TILES_PER_LAYER;
    const float* src = nullptr; int K0 = 0, N0 = 0, ld = 32;
    if (tile < 8)        { int c = tile >> 1, t = tile & 1;
                           src = eW1 + l * 257 * 32; K0 = c * 32; N0 = t * 16; ld = 32; }
    else if (tile < 16)  { int u = tile - 8; int c = u >> 1, t = u & 1;
                           src = eW1 + l * 257 * 32; K0 = 128 + c * 32; N0 = t * 16; ld = 32; }
    else if (tile < 18)  { int t = tile - 16; src = eW2 + l * 1024; K0 = 0; N0 = t * 16; ld = 32; }
    else if (tile < 20)  { int t = tile - 18; src = aW1 + l * 1024; K0 = 0; N0 = t * 16; ld = 32; }
    else if (tile < 22)  { int t = tile - 20; src = cW1 + l * 1024; K0 = 0; N0 = t * 16; ld = 32; }
    else if (tile < 62)  { int u = tile - 22; int c = u >> 3, t = u & 7;
                           src = sW1 + l * 160 * 128; K0 = c * 32; N0 = t * 16; ld = 128; }
    else                 { int u = tile - 62; int c = u >> 3, t = u & 7;
                           src = sW2 + l * 128 * 128; K0 = c * 32; N0 = t * 16; ld = 128; }
    int lane = threadIdx.x;
    int n = N0 + (lane & 15), hih = lane >> 4;
    _Float16* dst = wt + (size_t)gid * 512 + lane * 16;
#pragma unroll
    for (int h = 0; h < 16; ++h) {
        int k = K0 + kmap(h, hih);
        dst[h] = (_Float16)src[k * ld + n];
    }
}

// ------------------------------- embedding ---------------------------------
__global__ void k_embed(const float* pos, const float* sig, const float* cond,
                        const float* tsc,
                        const float* gW1, const float* gb1, const float* gW2, const float* gb2,
                        const float* cW1, const float* cb1, const float* cW2, const float* cb2,
                        const float* tW1, const float* tb1, const float* tW2, const float* tb2,
                        const float* pW, const float* pb,
                        float* s, _Float16* s_a, float* xw) {
    int i = blockIdx.x;            // global node 0..1023
    int b = i >> 9;
    int t = threadIdx.x;           // 0..127
    __shared__ float sh[64], ch[64], th[64], emb[192];
    if (t < 64) {
        float a = gb1[t] + sig[i * 2 + 0] * gW1[t] + sig[i * 2 + 1] * gW1[64 + t];
        sh[t] = silu(a);
        float c = cb1[t];
        for (int k = 0; k < 8; ++k) c += cond[b * 8 + k] * cW1[k * 64 + t];
        ch[t] = silu(c);
        th[t] = silu(tb1[t] + tsc[b] * tW1[t]);
    }
    __syncthreads();
    if (t < 64) {
        float a = gb2[t], c = cb2[t], d = tb2[t];
        for (int k = 0; k < 64; ++k) {
            a += sh[k] * gW2[k * 64 + t];
            c += ch[k] * cW2[k * 64 + t];
            d += th[k] * tW2[k * 64 + t];
        }
        emb[t] = a; emb[64 + t] = c; emb[128 + t] = d;
    }
    __syncthreads();
    float sv = pb[t];
    for (int k = 0; k < 192; ++k) sv += emb[k] * pW[k * 128 + t];
    s[i * 128 + t] = sv;
    // scatter into A-fragment layout (f16)
    int row = i & 15, ti = i >> 4;
    int c = t >> 5, k = t & 31;
    int hih = (k >> 3) & 1, h = ((k >> 4) << 3) | (k & 7);
    s_a[(size_t)ti * 2048 + c * 512 + (row + 16 * hih) * 16 + h] = (_Float16)sv;
    if (t < 3) xw[i * 3 + t] = pos[i * 3 + t];
}

// -------------------- hi/hj = s @ edgeW1 (per 16-node tile) ----------------
__global__ void k_pre(const _Float16* s_a, const _Float16* lwt,
                      float* hib, float* hjb) {
    int ti = blockIdx.x, lane = threadIdx.x;
    int n = lane & 15, hih = lane >> 4;
    V16H av[4];
#pragma unroll
    for (int c = 0; c < 4; ++c)
        av[c].v = *(const v16h*)(s_a + (size_t)ti * 2048 + c * 512 + lane * 16);
#pragma unroll
    for (int half = 0; half < 2; ++half) {
        float* out = half ? hjb : hib;
        int base = half ? 8 : 0;
#pragma unroll
        for (int t = 0; t < 2; ++t) {
            V8F acc;
#pragma unroll
            for (int r = 0; r < 8; ++r) acc.e[r] = 0.0f;
#pragma unroll
            for (int c = 0; c < 4; ++c) {
                V16H bv; bv.v = *(const v16h*)(lwt + (base + c * 2 + t) * 512 + lane * 16);
                acc.v = wmma16(av[c].v, bv.v, acc.v);
            }
#pragma unroll
            for (int r = 0; r < 8; ++r)
                out[(ti * 16 + r + 8 * hih) * 32 + t * 16 + n] = acc.e[r];
        }
    }
}

// ------------- fused edge / attention / aggregation (flash-style) ----------
__global__ void k_edge(const float* hib, const float* hjb, const float* xw,
                       const unsigned char* mask, const _Float16* lwt,
                       const float* distw, const float* eb1, const float* eb2,
                       const float* ab1, const float* ab2, const float* aW2,
                       const float* cb1, const float* cb2, const float* cW2,
                       _Float16* msa, float* dxb) {
    int i = blockIdx.x;           // global node (b*N + n)
    int b = i >> 9;
    int lane = threadIdx.x;
    int jl = lane & 15, hih = lane >> 4;
    const float NEGINF = -__builtin_inff();

    __shared__ __align__(16) float hjs[2][512];   // double-buffered TDM dest
    __shared__ float efs[16 * 32];
    __shared__ float tmp[16 * 32];
    __shared__ float d2s[16];
    __shared__ float dif[16][3];
    __shared__ float ps[16];
    __shared__ float aw2s[32], cw2s[32];
    aw2s[lane] = aW2[lane];
    cw2s[lane] = cW2[lane];

    int   kk[16];
    float hiv[16], dwv[16], eb1v[16];
#pragma unroll
    for (int h = 0; h < 16; ++h) {
        int k = kmap(h, hih);
        kk[h] = k;
        hiv[h]  = hib[i * 32 + k];
        dwv[h]  = distw[k];
        eb1v[h] = eb1[k];
    }
    float eb2a = eb2[jl],  eb2b = eb2[16 + jl];
    float ab1a = ab1[jl],  ab1b = ab1[16 + jl];
    float cb1a = cb1[jl],  cb1b = cb1[16 + jl];
    float ab2v = ab2[0],   cb2v = cb2[0];
    float xi0 = xw[i * 3], xi1 = xw[i * 3 + 1], xi2 = xw[i * 3 + 2];
    bool maski = mask[i] != 0;

    V16H bew[2], baw[2], bcw[2];
#pragma unroll
    for (int t = 0; t < 2; ++t) {
        bew[t].v = *(const v16h*)(lwt + (16 + t) * 512 + lane * 16);
        baw[t].v = *(const v16h*)(lwt + (18 + t) * 512 + lane * 16);
        bcw[t].v = *(const v16h*)(lwt + (20 + t) * 512 + lane * 16);
    }

    const unsigned lds_hj0 = (unsigned)(uintptr_t)(&hjs[0][0]);
    const unsigned lds_hj1 = (unsigned)(uintptr_t)(&hjs[1][0]);
    const float* hjbase = hjb + ((size_t)(b << 9)) * 32;

    float m = NEGINF, lsum = 0.0f, accms = 0.0f;
    float adx0 = 0.0f, adx1 = 0.0f, adx2 = 0.0f;

    // prologue: DMA tile 0 into buffer 0
    tdm_load_tile(lds_hj0, hjbase);

    for (int jt = 0; jt < 32; ++jt) {
        int j0g = (b << 9) + jt * 16;
        __syncthreads();               // prior LDS reads done before TDM reuse
        if (jt + 1 < 32) {
            // DMA next tile into the other buffer, wait for current only
            tdm_load_tile((jt & 1) ? lds_hj0 : lds_hj1,
                          hjbase + (size_t)(jt + 1) * 512);
            __builtin_amdgcn_s_wait_tensorcnt(1);
        } else {
            __builtin_amdgcn_s_wait_tensorcnt(0);
        }
        const float* hjt = hjs[jt & 1];
        if (lane < 16) {
            int jg = j0g + lane;
            float dx = xi0 - xw[jg * 3], dy = xi1 - xw[jg * 3 + 1], dz = xi2 - xw[jg * 3 + 2];
            dif[lane][0] = dx; dif[lane][1] = dy; dif[lane][2] = dz;
            d2s[lane] = dx * dx + dy * dy + dz * dz;
        }
        __syncthreads();
        float d2 = d2s[jl];

        // A fragment: h = silu(hi_i + hj_j + dist2 * w_d + b1), f16
        V16H af;
#pragma unroll
        for (int h = 0; h < 16; ++h) {
            float v = hiv[h] + hjt[jl * 32 + kk[h]] + d2 * dwv[h] + eb1v[h];
            af.e[h] = (_Float16)silu(v);
        }
        V8F e0, e1;
#pragma unroll
        for (int r = 0; r < 8; ++r) { e0.e[r] = 0.0f; e1.e[r] = 0.0f; }
        e0.v = wmma16(af.v, bew[0].v, e0.v);
        e1.v = wmma16(af.v, bew[1].v, e1.v);
#pragma unroll
        for (int r = 0; r < 8; ++r) {
            efs[(r + 8 * hih) * 32 + jl]      = e0.e[r] + eb2a;
            efs[(r + 8 * hih) * 32 + 16 + jl] = e1.e[r] + eb2b;
        }
        __syncthreads();

        // attention hidden = silu(ef @ attW1 + b1)
        V16H aef;
#pragma unroll
        for (int h = 0; h < 16; ++h) aef.e[h] = (_Float16)efs[jl * 32 + kk[h]];
        V8F a0, a1;
#pragma unroll
        for (int r = 0; r < 8; ++r) { a0.e[r] = 0.0f; a1.e[r] = 0.0f; }
        a0.v = wmma16(aef.v, baw[0].v, a0.v);
        a1.v = wmma16(aef.v, baw[1].v, a1.v);
#pragma unroll
        for (int r = 0; r < 8; ++r) {
            tmp[(r + 8 * hih) * 32 + jl]      = silu(a0.e[r] + ab1a);
            tmp[(r + 8 * hih) * 32 + 16 + jl] = silu(a1.e[r] + ab1b);
        }
        __syncthreads();
        float lg = ab2v;
        for (int n2 = 0; n2 < 32; ++n2) lg += tmp[jl * 32 + n2] * aw2s[n2];
        bool valid = maski && (mask[j0g + jl] != 0) && (d2 <= MAXD2);
        if (!valid) lg = NEGINF;
        __syncthreads();

        // coord weight hidden = silu(ef @ coordW1 + b1)  (reuse tmp)
        V8F c0, c1;
#pragma unroll
        for (int r = 0; r < 8; ++r) { c0.e[r] = 0.0f; c1.e[r] = 0.0f; }
        c0.v = wmma16(aef.v, bcw[0].v, c0.v);
        c1.v = wmma16(aef.v, bcw[1].v, c1.v);
#pragma unroll
        for (int r = 0; r < 8; ++r) {
            tmp[(r + 8 * hih) * 32 + jl]      = silu(c0.e[r] + cb1a);
            tmp[(r + 8 * hih) * 32 + 16 + jl] = silu(c1.e[r] + cb1b);
        }
        __syncthreads();
        float cwv = cb2v;
        for (int n2 = 0; n2 < 32; ++n2) cwv += tmp[jl * 32 + n2] * cw2s[n2];

        // online softmax update
        float tmax = lg;
#pragma unroll
        for (int off = 16; off; off >>= 1) tmax = fmaxf(tmax, __shfl_xor(tmax, off, 32));
        float nm = fmaxf(m, tmax);
        float scale, p;
        if (nm == NEGINF) { scale = 1.0f; p = 0.0f; }
        else {
            scale = (m == NEGINF) ? 0.0f : __expf(m - nm);
            p = (lg == NEGINF) ? 0.0f : __expf(lg - nm);
        }
        m = nm;
        float pme = (lane < 16) ? p : 0.0f;
        float psum = pme;
#pragma unroll
        for (int off = 16; off; off >>= 1) psum += __shfl_xor(psum, off, 32);
        lsum = lsum * scale + psum;
        float w = pme * cwv;
        adx0 = adx0 * scale + w * dif[jl][0];
        adx1 = adx1 * scale + w * dif[jl][1];
        adx2 = adx2 * scale + w * dif[jl][2];
        if (lane < 16) ps[jl] = p;
        __syncthreads();
        float acc2 = 0.0f;
#pragma unroll
        for (int jj = 0; jj < 16; ++jj) acc2 += ps[jj] * efs[jj * 32 + lane];
        accms = accms * scale + acc2;
    }

    float invl = (lsum > 0.0f) ? (1.0f / lsum) : 0.0f;
    // m_s written directly in A-fragment layout (feature = lane)
    {
        int n2 = lane, ti = i >> 4, row = i & 15;
        int hih2 = (n2 >> 3) & 1, h2 = ((n2 >> 4) << 3) | (n2 & 7);
        msa[(size_t)ti * 512 + (row + 16 * hih2) * 16 + h2] = (_Float16)(accms * invl);
    }
#pragma unroll
    for (int off = 16; off; off >>= 1) {
        adx0 += __shfl_xor(adx0, off, 32);
        adx1 += __shfl_xor(adx1, off, 32);
        adx2 += __shfl_xor(adx2, off, 32);
    }
    if (lane == 0) {
        dxb[i * 3 + 0] = adx0 * invl;
        dxb[i * 3 + 1] = adx1 * invl;
        dxb[i * 3 + 2] = adx2 * invl;
    }
}

// ----------------- scalar residual MLP + coordinate update -----------------
__global__ void k_scalar(_Float16* s_a, const _Float16* msa, const _Float16* lwt,
                         const float* sb1, const float* sb2,
                         float* s, float* xw, const float* dxb) {
    int ti = blockIdx.x, lane = threadIdx.x;
    int n = lane & 15, hih = lane >> 4;
    __shared__ float hid[16 * 128];
    __shared__ float sn[16 * 128];
    int kk[16];
#pragma unroll
    for (int h = 0; h < 16; ++h) kk[h] = kmap(h, hih);

    V16H av[5];
#pragma unroll
    for (int c = 0; c < 4; ++c)
        av[c].v = *(const v16h*)(s_a + (size_t)ti * 2048 + c * 512 + lane * 16);
    av[4].v = *(const v16h*)(msa + (size_t)ti * 512 + lane * 16);

    // hidden = silu([s, m_s] @ W1 + b1)
#pragma unroll
    for (int t = 0; t < 8; ++t) {
        V8F acc;
#pragma unroll
        for (int r = 0; r < 8; ++r) acc.e[r] = 0.0f;
#pragma unroll
        for (int c = 0; c < 5; ++c) {
            V16H bv; bv.v = *(const v16h*)(lwt + (22 + c * 8 + t) * 512 + lane * 16);
            acc.v = wmma16(av[c].v, bv.v, acc.v);
        }
        float b1v = sb1[t * 16 + n];
#pragma unroll
        for (int r = 0; r < 8; ++r)
            hid[(r + 8 * hih) * 128 + t * 16 + n] = silu(acc.e[r] + b1v);
    }
    __syncthreads();

    V16H ah[4];
#pragma unroll
    for (int c = 0; c < 4; ++c)
#pragma unroll
        for (int h = 0; h < 16; ++h)
            ah[c].e[h] = (_Float16)hid[(lane & 15) * 128 + c * 32 + kk[h]];

    // s += hidden @ W2 + b2
#pragma unroll
    for (int t = 0; t < 8; ++t) {
        V8F acc;
#pragma unroll
        for (int r = 0; r < 8; ++r) acc.e[r] = 0.0f;
#pragma unroll
        for (int c = 0; c < 4; ++c) {
            V16H bv; bv.v = *(const v16h*)(lwt + (62 + c * 8 + t) * 512 + lane * 16);
            acc.v = wmma16(ah[c].v, bv.v, acc.v);
        }
        float b2v = sb2[t * 16 + n];
#pragma unroll
        for (int r = 0; r < 8; ++r) {
            int row = r + 8 * hih, node = ti * 16 + row;
            float v = s[node * 128 + t * 16 + n] + acc.e[r] + b2v;
            s[node * 128 + t * 16 + n] = v;
            sn[row * 128 + t * 16 + n] = v;
        }
    }
    __syncthreads();

    // refresh A-fragment copy of s
#pragma unroll
    for (int c = 0; c < 4; ++c) {
        _Float16* dst = s_a + (size_t)ti * 2048 + c * 512 + lane * 16;
#pragma unroll
        for (int h = 0; h < 16; ++h)
            dst[h] = (_Float16)sn[(lane & 15) * 128 + c * 32 + kk[h]];
    }
    // x += dx
    if (lane < 16) {
        int node = ti * 16 + lane;
        for (int c = 0; c < 3; ++c) xw[node * 3 + c] += dxb[node * 3 + c];
    }
}

// ------------------------------ output proj --------------------------------
__global__ void k_out(const float* s, const float* oW, const float* ob, float* out) {
    int idx = blockIdx.x * blockDim.x + threadIdx.x;
    if (idx >= TOTN * 2) return;
    int i = idx >> 1, o = idx & 1;
    float v = ob[o];
    for (int k = 0; k < 128; ++k) v += s[i * 128 + k] * oW[k * 2 + o];
    out[idx] = v;
}

// ---------------------------------------------------------------------------
// d_in order: jax.tree_leaves of the setup_inputs() dict (keys sorted at each
// nesting level):
//  0 cond | 1 dom_positions | 2 dom_signals | 3 mask
//  4..7   params.cond   {W1,W2,b1,b2}
//  8..11  layers.att    {W1,W2,b1,b2}   (leading dim L=8)
//  12..15 layers.coord  {W1,W2,b1,b2}
//  16..19 layers.edge   {W1,W2,b1,b2}
//  20..23 layers.scalar {W1,W2,b1,b2}
//  24 out_W | 25 out_b | 26 proj_W | 27 proj_b
//  28..31 params.sig {W1,W2,b1,b2} | 32..35 params.time {W1,W2,b1,b2}
//  36 t_scalar
// ---------------------------------------------------------------------------
extern "C" void kernel_launch(void* const* d_in, const int* in_sizes, int n_in,
                              void* d_out, int out_size, void* d_ws, size_t ws_size,
                              hipStream_t stream) {
    const float* condv = (const float*)d_in[0];
    const float* pos   = (const float*)d_in[1];
    const float* sig   = (const float*)d_in[2];
    const unsigned char* mask = (const unsigned char*)d_in[3];
    const float* cW1 = (const float*)d_in[4],  *cW2 = (const float*)d_in[5];
    const float* cb1 = (const float*)d_in[6],  *cb2 = (const float*)d_in[7];
    const float* aW1 = (const float*)d_in[8],  *aW2 = (const float*)d_in[9];
    const float* ab1 = (const float*)d_in[10], *ab2 = (const float*)d_in[11];
    const float* oW1 = (const float*)d_in[12], *oW2 = (const float*)d_in[13]; // coord
    const float* ob1 = (const float*)d_in[14], *ob2 = (const float*)d_in[15];
    const float* eW1 = (const float*)d_in[16], *eW2 = (const float*)d_in[17];
    const float* eb1 = (const float*)d_in[18], *eb2 = (const float*)d_in[19];
    const float* sW1 = (const float*)d_in[20], *sW2 = (const float*)d_in[21];
    const float* sb1 = (const float*)d_in[22], *sb2 = (const float*)d_in[23];
    const float* outW = (const float*)d_in[24], *outb = (const float*)d_in[25];
    const float* pW  = (const float*)d_in[26], *pb  = (const float*)d_in[27];
    const float* gW1 = (const float*)d_in[28], *gW2 = (const float*)d_in[29];
    const float* gb1 = (const float*)d_in[30], *gb2 = (const float*)d_in[31];
    const float* tW1 = (const float*)d_in[32], *tW2 = (const float*)d_in[33];
    const float* tb1 = (const float*)d_in[34], *tb2 = (const float*)d_in[35];
    const float* tsc = (const float*)d_in[36];

    char* ws = (char*)d_ws;
    _Float16* wt  = (_Float16*)(ws + 0);         // 8*94*1024B  = 770048
    float*    sbf = (float*)(ws + 770048);       // 1024*128*4  = 524288
    _Float16* s_a = (_Float16*)(ws + 1294336);   // 64*2048*2   = 262144
    _Float16* msa = (_Float16*)(ws + 1556480);   // 64*512*2    = 65536
    float*    hib = (float*)(ws + 1622016);      // 1024*32*4   = 131072
    float*    hjb = (float*)(ws + 1753088);      // 131072
    float*    xw  = (float*)(ws + 1884160);      // 12288
    float*    dxb = (float*)(ws + 1896448);      // 12288      (total 1908736 B)

    k_prep_weights<<<NLAY * TILES_PER_LAYER, 32, 0, stream>>>(
        eW1, eW2, aW1, oW1, sW1, sW2, wt);

    k_embed<<<TOTN, 128, 0, stream>>>(
        pos, sig, condv, tsc,
        gW1, gb1, gW2, gb2, cW1, cb1, cW2, cb2, tW1, tb1, tW2, tb2,
        pW, pb, sbf, s_a, xw);

    for (int l = 0; l < NLAY; ++l) {
        const _Float16* lwt = wt + (size_t)l * TILES_PER_LAYER * 512;
        k_pre<<<TOTN / 16, 32, 0, stream>>>(s_a, lwt, hib, hjb);
        k_edge<<<TOTN, 32, 0, stream>>>(
            hib, hjb, xw, mask, lwt,
            eW1 + l * 257 * 32 + 256 * 32,     // dist2 weight row of edge W1
            eb1 + l * 32, eb2 + l * 32,
            ab1 + l * 32, ab2 + l, aW2 + l * 32,
            ob1 + l * 32, ob2 + l, oW2 + l * 32,
            msa, dxb);
        k_scalar<<<TOTN / 16, 32, 0, stream>>>(
            s_a, msa, lwt, sb1 + l * 128, sb2 + l * 128, sbf, xw, dxb);
    }

    k_out<<<(TOTN * 2 + 255) / 256, 256, 0, stream>>>(sbf, outW, outb, (float*)d_out);
}
